// TransformerEncoderLayer_58926951301696
// MI455X (gfx1250) — matrix operand order repaired and checked by reading, not compiled, not run on hardware
//
#include <hip/hip_runtime.h>

// ---------------------------------------------------------------------------
// Transformer encoder layer forward, MI455X (gfx1250, wave32, WMMA + TDM).
// All matmuls via v_wmma_f32_16x16x32_f16 (f16 inputs, fp32 accumulate).
// GEMM B-tiles are staged into LDS by the Tensor Data Mover (tensor_load_to_lds
// + s_wait_tensorcnt). Attention is flash-style fused; softmax row-sums are
// computed on the tensor core (P x ones) instead of lane shuffles.
// ---------------------------------------------------------------------------

typedef _Float16 half_t;
typedef __attribute__((ext_vector_type(16))) _Float16 v16h;
typedef __attribute__((ext_vector_type(8)))  _Float16 v8h;
typedef __attribute__((ext_vector_type(8)))  float    v8f;
typedef __attribute__((ext_vector_type(4)))  float    v4f;
typedef __attribute__((ext_vector_type(4)))  unsigned int v4u;
typedef __attribute__((ext_vector_type(8)))  int      v8i;
typedef __attribute__((ext_vector_type(4)))  int      v4i;

constexpr int Bc = 2, Sc = 2048, Dc = 768, Hc = 12, DHc = 64, Fc = 3072;
constexpr int Mrows = Bc * Sc;   // 4096

__device__ __forceinline__ v16h cat8(v8h lo, v8h hi) {
  v16h r;
  ((v8h*)&r)[0] = lo;
  ((v8h*)&r)[1] = hi;
  return r;
}

__device__ __forceinline__ v8f wmma16(v16h a, v16h b, v8f c) {
  // D = A(16x32,f16) * B(32x16,f16) + C(16x16,f32)
  return __builtin_amdgcn_wmma_f32_16x16x32_f16(
      /*neg_a=*/false, a, /*neg_b=*/false, b,
      /*c_mod=*/(short)0, c, /*reuse_a=*/false, /*reuse_b=*/false);
}

#if __has_builtin(__builtin_amdgcn_tensor_load_to_lds)
#define USE_TDM 1
#else
#define USE_TDM 0
#endif

#if USE_TDM
// Issue a TDM load of a 2-D f16 tile [rows x 32 elems] (row stride = K elems)
// from global into LDS at byte offset lds_addr. One call per wave; completion
// tracked with TENSORcnt. D# layout per CDNA5 ISA.
__device__ __forceinline__ void tdm_load_tile_f16_32(
    const half_t* gsrc, unsigned int lds_addr, int rows, int k_stride) {
  unsigned long long ga = (unsigned long long)(uintptr_t)gsrc;
  v4u g0;
  g0[0] = 1u;                                   // count=1 (valid user D#)
  g0[1] = lds_addr;                             // lds_addr (bytes)
  g0[2] = (unsigned int)ga;                     // global_addr[31:0]
  g0[3] = (unsigned int)((ga >> 32) & 0x01FFFFFFu) | (2u << 30);  // addr[56:32], type=2
  v8i g1;
  unsigned int K = (unsigned int)k_stride;
  unsigned int R = (unsigned int)rows;
  g1[0] = (int)(1u << 16);                      // wg_mask=0, data_size=1 (2 bytes)
  g1[1] = (int)(32u << 16);                     // tensor_dim0[15:0]=32 at bits[63:48]
  g1[2] = (int)((R & 0xFFFFu) << 16);           // dim0 hi=0, tensor_dim1[15:0]=rows
  g1[3] = (int)(32u << 16);                     // dim1 hi=0, tile_dim0=32
  g1[4] = (int)R;                               // tile_dim1=rows, tile_dim2=0
  g1[5] = (int)K;                               // tensor_dim0_stride[31:0]=K
  g1[6] = 0;                                    // stride hi, tensor_dim1_stride lo
  g1[7] = 0;
  v4i gz = {};
#if __clang_major__ >= 23
  v8i gz8 = {};
  __builtin_amdgcn_tensor_load_to_lds(g0, g1, gz, gz, gz8, 0);
#else
  __builtin_amdgcn_tensor_load_to_lds(g0, g1, gz, gz, 0);
#endif
}
#endif

// ---------------------------------------------------------------------------
// Weight pack kernels: produce f16 "B-transposed" layouts [N][K].
// ---------------------------------------------------------------------------

// Wq/Wk/Wv: (H, D, DH) fp32 -> Wt[j][d] f16 with j = h*DH+e  (N=768, K=768)
__global__ __launch_bounds__(256) void pack_headw_kernel(
    const float* __restrict__ W, half_t* __restrict__ Wt) {
  int idx = blockIdx.x * 256 + threadIdx.x;    // over 768*768
  int j = idx / Dc;
  int d = idx - j * Dc;
  int h = j >> 6, e = j & 63;
  Wt[idx] = (half_t)W[((size_t)h * Dc + d) * DHc + e];
}

// W (K x N) fp32 row-major -> Wt (N x K) f16
__global__ __launch_bounds__(256) void pack_T_kernel(
    const float* __restrict__ W, half_t* __restrict__ Wt, int N, int K) {
  size_t idx = (size_t)blockIdx.x * 256 + threadIdx.x;  // over N*K
  int n = (int)(idx / K);
  int k = (int)(idx - (size_t)n * K);
  Wt[idx] = (half_t)W[(size_t)k * N + n];
}

// ---------------------------------------------------------------------------
// Generic WMMA GEMM:  out = act( A[MxK] * Bt[NxK]^T + bias + resid )
// 256 threads = 8 waves (2x4); block tile 64x256; wave tile 32x64
// (2x4 C tiles -> 8 WMMAs per 32-wide K step). B tile staged by the TDM.
// OUT_MODE: 0 = fp32 [M][N], 3 = f16 [M][N],
//           1 = f16 [b,h,s,e] (Q/K layout), 2 = f16 [b,h,e,s] (V^T layout)
// ---------------------------------------------------------------------------
template <int A_HALF, int OUT_MODE>
__global__ __launch_bounds__(256) void gemm_wmma_kernel(
    const void* __restrict__ Ap, const half_t* __restrict__ Bt,
    const float* __restrict__ bias, const float* __restrict__ resid,
    void* __restrict__ outp, int M, int N, int K, int relu) {
  __shared__ half_t As[64 * 32];    // [row][k]
  __shared__ half_t Bs[256 * 32];   // [n][k]

  const int tid  = threadIdx.x;
  const int lane = tid & 31;
  const int wave = tid >> 5;
  const int wm   = wave & 1;        // 2 waves along M
  const int wn   = wave >> 1;       // 4 waves along N
  const int mb   = blockIdx.x * 64;
  const int nb   = blockIdx.y * 256;
  const int l15  = lane & 15;
  const int hi   = lane >> 4;       // which 16-lane half

  const int arow = tid >> 2;        // 0..63
  const int acg  = (tid & 3) * 8;   // 0,8,16,24

  v8f acc[2][4] = {};

  const half_t* Bbase = Bt + (size_t)nb * K;
#if USE_TDM
  const unsigned int bs_lds = (unsigned int)(uintptr_t)(void*)Bs;
#endif

  for (int k0 = 0; k0 < K; k0 += 32) {
    __syncthreads();   // previous tile fully consumed by all waves
#if USE_TDM
    if (wave == 0) tdm_load_tile_f16_32(Bbase + k0, bs_lds, 256, K);
#else
    {
      const half_t* p = Bt + (size_t)(nb + tid) * K + k0;
#pragma unroll
      for (int c = 0; c < 4; ++c)
        *(v8h*)(Bs + tid * 32 + c * 8) = *(const v8h*)(p + c * 8);
    }
#endif
    // ---- stage A tile (64x32) into LDS as f16 (needs convert -> VALU) ----
    if (A_HALF) {
      const half_t* A = (const half_t*)Ap;
      *(v8h*)(As + arow * 32 + acg) =
          *(const v8h*)(A + (size_t)(mb + arow) * K + k0 + acg);
      __builtin_prefetch((const half_t*)Ap + (size_t)(mb + arow) * K + k0 + 32 + acg, 0, 1);
    } else {
      const float* A = (const float*)Ap;
      const float* p = A + (size_t)(mb + arow) * K + k0 + acg;
      v4f f0 = *(const v4f*)p;
      v4f f1 = *(const v4f*)(p + 4);
      v8h v;
      v[0] = (half_t)f0[0]; v[1] = (half_t)f0[1];
      v[2] = (half_t)f0[2]; v[3] = (half_t)f0[3];
      v[4] = (half_t)f1[0]; v[5] = (half_t)f1[1];
      v[6] = (half_t)f1[2]; v[7] = (half_t)f1[3];
      *(v8h*)(As + arow * 32 + acg) = v;
      __builtin_prefetch(p + 32, 0, 1);
    }
#if USE_TDM
    if (wave == 0) __builtin_amdgcn_s_wait_tensorcnt(0);
#endif
    __syncthreads();   // B tile (TDM) + A tile (ds) visible to all waves

    // ---- fragments per ISA VGPR layouts ----
    v16h af[2], bf[4];
#pragma unroll
    for (int i = 0; i < 2; ++i) {
      // A 16x32: lane<16 -> M=lane, K in {kb..kb+7, kb+16..kb+23}, kb=8*hi
      const half_t* p = As + (wm * 32 + i * 16 + l15) * 32 + hi * 8;
      af[i] = cat8(*(const v8h*)p, *(const v8h*)(p + 16));
    }
#pragma unroll
    for (int j = 0; j < 4; ++j) {
      // B 32x16: lane<16 -> N=lane, K in {kb..kb+7, kb+16..kb+23}, kb=8*hi
      const half_t* p = Bs + (wn * 64 + j * 16 + l15) * 32 + hi * 8;
      bf[j] = cat8(*(const v8h*)p, *(const v8h*)(p + 16));
    }
#pragma unroll
    for (int i = 0; i < 2; ++i)
#pragma unroll
      for (int j = 0; j < 4; ++j)
        acc[i][j] = wmma16(af[i], bf[j], acc[i][j]);
  }

  // ---- epilogue: C layout VGPR r, half -> M = r + 8*hi ----
#pragma unroll
  for (int i = 0; i < 2; ++i) {
#pragma unroll
    for (int j = 0; j < 4; ++j) {
#pragma unroll
      for (int r = 0; r < 8; ++r) {
        int m = mb + wm * 32 + i * 16 + r + (hi << 3);
        int n = nb + wn * 64 + j * 16 + l15;
        float v = acc[i][j][r];
        if (bias)  v += bias[n];
        if (resid) v += resid[(size_t)m * N + n];
        if (relu)  v = fmaxf(v, 0.0f);
        if (OUT_MODE == 0) {
          ((float*)outp)[(size_t)m * N + n] = v;
        } else if (OUT_MODE == 3) {
          ((half_t*)outp)[(size_t)m * N + n] = (half_t)v;
        } else {
          int bb = m >> 11, s = m & (Sc - 1);  // S = 2048
          int hh = n >> 6,  e = n & 63;        // DH = 64
          size_t off = (OUT_MODE == 1)
              ? ((((size_t)bb * Hc + hh) * Sc + s) * DHc + e)    // [b,h,s,e]
              : ((((size_t)bb * Hc + hh) * DHc + e) * Sc + s);   // [b,h,e,s]
          ((half_t*)outp)[off] = (half_t)v;
        }
      }
    }
  }
}

// ---------------------------------------------------------------------------
// Flash attention: one wave per 16-row Q tile, key blocks of 32.
// Q/K f16 [b,h,s,e]; V f16 transposed [b,h,e,s]; ctx fp32 [b*s][h*64+e].
// Row sums of P are computed on the tensor core: P(16x32) x ones(32x16)
// puts the row sum into every column of a C tile (no lane shuffles).
// ---------------------------------------------------------------------------
__global__ __launch_bounds__(128) void flash_attn_kernel(
    const half_t* __restrict__ Qh, const half_t* __restrict__ Kh,
    const half_t* __restrict__ Vth, float* __restrict__ ctx) {
  __shared__ half_t Pl[4][16 * 32];   // per-wave P tile (16 q-rows x 32 keys)

  const int lane = threadIdx.x & 31;
  const int wave = threadIdx.x >> 5;
  const int l15  = lane & 15;
  const int hi   = lane >> 4;
  const int bh   = blockIdx.y;          // 0..23
  const int b    = bh / Hc;
  const int h    = bh - b * Hc;
  const int m0   = (blockIdx.x * 4 + wave) * 16;

  const half_t* Qbase = Qh  + ((size_t)bh * Sc + m0) * DHc;
  const half_t* Kbase = Kh  + (size_t)bh * Sc * DHc;
  const half_t* Vbase = Vth + (size_t)bh * DHc * Sc;

  // Q A-fragments (16x32 each, covering e=[0,32) and [32,64)), pre-scaled 1/8
  v16h aq[2];
#pragma unroll
  for (int t = 0; t < 2; ++t) {
    const half_t* p = Qbase + (size_t)l15 * DHc + t * 32 + hi * 8;
    v8h lo = *(const v8h*)p;
    v8h hh = *(const v8h*)(p + 16);
    lo = lo * (half_t)0.125f;   // 1/sqrt(DH), exact power of two
    hh = hh * (half_t)0.125f;
    aq[t] = cat8(lo, hh);
  }

  v16h vones;
#pragma unroll
  for (int i = 0; i < 16; ++i) vones[i] = (half_t)1.0f;

  float mrow[8], lrow[8];
#pragma unroll
  for (int r = 0; r < 8; ++r) { mrow[r] = -3.0e38f; lrow[r] = 0.0f; }
  v8f cacc[4] = {};   // 16x64 fp32 context accumulator (4 C tiles)

  for (int j = 0; j < Sc; j += 32) {
    // ---- scores: 2 key-column groups x 2 e-halves = 4 WMMAs ----
    v8f sc[2];
#pragma unroll
    for (int g = 0; g < 2; ++g) {
      v8f z = {};
#pragma unroll
      for (int t = 0; t < 2; ++t) {
        const half_t* p = Kbase + (size_t)(j + g * 16 + l15) * DHc + t * 32 + hi * 8;
        v16h bk = cat8(*(const v8h*)p, *(const v8h*)(p + 16));
        z = wmma16(aq[t], bk, z);
      }
      sc[g] = z;
    }

    // ---- online softmax max: row lives in one 16-lane half of the wave ----
    float bm[8];
#pragma unroll
    for (int r = 0; r < 8; ++r) bm[r] = fmaxf(sc[0][r], sc[1][r]);
#pragma unroll
    for (int mask = 1; mask < 16; mask <<= 1)
#pragma unroll
      for (int r = 0; r < 8; ++r) bm[r] = fmaxf(bm[r], __shfl_xor(bm[r], mask, 16));

    float corr[8];
#pragma unroll
    for (int r = 0; r < 8; ++r) {
      float mn = fmaxf(mrow[r], bm[r]);
      corr[r]  = __expf(mrow[r] - mn);
      mrow[r]  = mn;
      float p0 = __expf(sc[0][r] - mn);
      float p1 = __expf(sc[1][r] - mn);
      int prow = r + (hi << 3);
      Pl[wave][prow * 32 + l15]      = (half_t)p0;   // C-layout -> LDS row-major
      Pl[wave][prow * 32 + 16 + l15] = (half_t)p1;
    }
#pragma unroll
    for (int c = 0; c < 4; ++c)
#pragma unroll
      for (int r = 0; r < 8; ++r) cacc[c][r] *= corr[r];

    __syncthreads();  // uniform trip count; orders P store -> A-frag reload

    // reload P as a 16x32 A-fragment
    v16h ap;
    {
      const half_t* p = &Pl[wave][l15 * 32 + hi * 8];
      ap = cat8(*(const v8h*)p, *(const v8h*)(p + 16));
    }
    // ---- row sums on the tensor core: every column of sums = rowsum(P) ----
    {
      v8f z = {};
      v8f sums = wmma16(ap, vones, z);
#pragma unroll
      for (int r = 0; r < 8; ++r) lrow[r] = lrow[r] * corr[r] + sums[r];
    }
    // ---- ctx += P(16x32) * V(32x64): 4 WMMAs (V^T rows are contiguous) ----
#pragma unroll
    for (int c = 0; c < 4; ++c) {
      const half_t* p = Vbase + (size_t)(c * 16 + l15) * Sc + j + hi * 8;
      v16h bv = cat8(*(const v8h*)p, *(const v8h*)(p + 16));
      cacc[c] = wmma16(ap, bv, cacc[c]);
    }
    __syncthreads();
  }

  // ---- normalize and write ctx fp32 in concat-heads layout [b*s][h*64+e] ----
#pragma unroll
  for (int c = 0; c < 4; ++c)
#pragma unroll
    for (int r = 0; r < 8; ++r) {
      float v  = cacc[c][r] / lrow[r];
      int row  = m0 + r + (hi << 3);
      int col  = h * DHc + c * 16 + l15;
      ctx[((size_t)b * Sc + row) * Dc + col] = v;
    }
}

// ---------------------------------------------------------------------------
// Row LayerNorm over D=768, one 256-thread block per row.
// ---------------------------------------------------------------------------
__global__ __launch_bounds__(256) void layernorm_kernel(
    const float* __restrict__ in, const float* __restrict__ g,
    const float* __restrict__ be, float* __restrict__ out) {
  __shared__ float red[8];
  const int tid  = threadIdx.x;
  const int lane = tid & 31;
  const int wave = tid >> 5;
  const float* x = in + (size_t)blockIdx.x * Dc;

  float s = 0.0f;
  for (int i = tid; i < Dc; i += 256) s += x[i];
#pragma unroll
  for (int mask = 1; mask < 32; mask <<= 1) s += __shfl_xor(s, mask, 32);
  if (lane == 0) red[wave] = s;
  __syncthreads();
  float tot = 0.0f;
#pragma unroll
  for (int k = 0; k < 8; ++k) tot += red[k];
  const float mu = tot * (1.0f / Dc);
  __syncthreads();

  float vs = 0.0f;
  for (int i = tid; i < Dc; i += 256) { float d = x[i] - mu; vs += d * d; }
#pragma unroll
  for (int mask = 1; mask < 32; mask <<= 1) vs += __shfl_xor(vs, mask, 32);
  if (lane == 0) red[wave] = vs;
  __syncthreads();
  float vtot = 0.0f;
#pragma unroll
  for (int k = 0; k < 8; ++k) vtot += red[k];
  const float inv = rsqrtf(vtot * (1.0f / Dc) + 1e-5f);

  float* o = out + (size_t)blockIdx.x * Dc;
  for (int i = tid; i < Dc; i += 256) o[i] = (x[i] - mu) * inv * g[i] + be[i];
}

// ---------------------------------------------------------------------------
// Host-side orchestration.
// ---------------------------------------------------------------------------
extern "C" void kernel_launch(void* const* d_in, const int* in_sizes, int n_in,
                              void* d_out, int out_size, void* d_ws, size_t ws_size,
                              hipStream_t stream) {
  (void)in_sizes; (void)n_in; (void)out_size; (void)ws_size;

  const float* x   = (const float*)d_in[0];
  const float* Wq  = (const float*)d_in[1];
  const float* bq  = (const float*)d_in[2];
  const float* Wk  = (const float*)d_in[3];
  const float* bk  = (const float*)d_in[4];
  const float* Wv  = (const float*)d_in[5];
  const float* bv  = (const float*)d_in[6];
  const float* Wo  = (const float*)d_in[7];
  const float* bo  = (const float*)d_in[8];
  const float* g1  = (const float*)d_in[9];
  const float* be1 = (const float*)d_in[10];
  const float* W1  = (const float*)d_in[11];
  const float* b1  = (const float*)d_in[12];
  const float* W2  = (const float*)d_in[13];
  const float* b2  = (const float*)d_in[14];
  const float* g2  = (const float*)d_in[15];
  const float* be2 = (const float*)d_in[16];

  // workspace carve-out (all offsets 256B aligned)
  char* w = (char*)d_ws;
  auto alloc = [&](size_t bytes) -> void* {
    void* p = (void*)w;
    w += (bytes + 255) & ~(size_t)255;
    return p;
  };
  half_t* WqT  = (half_t*)alloc((size_t)Dc * Dc * sizeof(half_t));
  half_t* WkT  = (half_t*)alloc((size_t)Dc * Dc * sizeof(half_t));
  half_t* WvT  = (half_t*)alloc((size_t)Dc * Dc * sizeof(half_t));
  half_t* WoT  = (half_t*)alloc((size_t)Dc * Dc * sizeof(half_t));
  half_t* W1T  = (half_t*)alloc((size_t)Fc * Dc * sizeof(half_t));
  half_t* W2T  = (half_t*)alloc((size_t)Dc * Fc * sizeof(half_t));
  half_t* Qh   = (half_t*)alloc((size_t)Bc * Hc * Sc * DHc * sizeof(half_t));
  half_t* Kh   = (half_t*)alloc((size_t)Bc * Hc * Sc * DHc * sizeof(half_t));
  half_t* Vth  = (half_t*)alloc((size_t)Bc * Hc * DHc * Sc * sizeof(half_t));
  float*  ctx  = (float*)alloc((size_t)Mrows * Dc * sizeof(float));
  float*  y1   = (float*)alloc((size_t)Mrows * Dc * sizeof(float));
  float*  x1   = (float*)alloc((size_t)Mrows * Dc * sizeof(float));
  half_t* hbuf = (half_t*)alloc((size_t)Mrows * Fc * sizeof(half_t));
  float*  y2   = (float*)alloc((size_t)Mrows * Dc * sizeof(float));

  // 1) pack weights to f16 [N][K]
  pack_headw_kernel<<<(Dc * Dc) / 256, 256, 0, stream>>>(Wq, WqT);
  pack_headw_kernel<<<(Dc * Dc) / 256, 256, 0, stream>>>(Wk, WkT);
  pack_headw_kernel<<<(Dc * Dc) / 256, 256, 0, stream>>>(Wv, WvT);
  pack_T_kernel<<<(Dc * Dc) / 256, 256, 0, stream>>>(Wo, WoT, Dc, Dc);
  pack_T_kernel<<<(Fc * Dc) / 256, 256, 0, stream>>>(W1, W1T, Fc, Dc);
  pack_T_kernel<<<(Dc * Fc) / 256, 256, 0, stream>>>(W2, W2T, Dc, Fc);

  const dim3 gemm768(Mrows / 64, Dc / 256);    // 64 x 3
  const dim3 gemm3072(Mrows / 64, Fc / 256);   // 64 x 12

  // 2) QKV projections (A = x fp32, out f16 in attention layouts)
  gemm_wmma_kernel<0, 1><<<gemm768, 256, 0, stream>>>(x, WqT, bq, nullptr, Qh,
                                                      Mrows, Dc, Dc, 0);
  gemm_wmma_kernel<0, 1><<<gemm768, 256, 0, stream>>>(x, WkT, bk, nullptr, Kh,
                                                      Mrows, Dc, Dc, 0);
  gemm_wmma_kernel<0, 2><<<gemm768, 256, 0, stream>>>(x, WvT, bv, nullptr, Vth,
                                                      Mrows, Dc, Dc, 0);

  // 3) fused attention -> ctx (concat-heads fp32)
  flash_attn_kernel<<<dim3(Sc / 64, Bc * Hc), 128, 0, stream>>>(Qh, Kh, Vth, ctx);

  // 4) output projection + residual(x)
  gemm_wmma_kernel<0, 0><<<gemm768, 256, 0, stream>>>(ctx, WoT, bo, x, y1,
                                                      Mrows, Dc, Dc, 0);
  // 5) LayerNorm 1 -> x1
  layernorm_kernel<<<Mrows, 256, 0, stream>>>(y1, g1, be1, x1);

  // 6) FFN1: relu(x1 @ W1 + b1) -> f16
  gemm_wmma_kernel<0, 3><<<gemm3072, 256, 0, stream>>>(x1, W1T, b1, nullptr, hbuf,
                                                       Mrows, Fc, Dc, 1);
  // 7) FFN2: hbuf @ W2 + b2 + residual(x1)
  gemm_wmma_kernel<1, 0><<<gemm768, 256, 0, stream>>>(hbuf, W2T, b2, x1, y2,
                                                      Mrows, Dc, Fc, 0);
  // 8) LayerNorm 2 -> output
  layernorm_kernel<<<Mrows, 256, 0, stream>>>(y2, g2, be2, (float*)d_out);
}
